// GCN_7576322311022
// MI455X (gfx1250) — matrix operand order repaired and checked by reading, not compiled, not run on hardware
//
#include <hip/hip_runtime.h>

typedef __attribute__((ext_vector_type(16))) __bf16 v16bf;
typedef __attribute__((ext_vector_type(8)))  float  v8f;

#define F_IN  512
#define F_HID 64
#define F_OUT 10

// ---------------- degree / norm ----------------
__global__ void k_init_deg(float* __restrict__ deg, int n) {
    int i = blockIdx.x * blockDim.x + threadIdx.x;
    if (i < n) deg[i] = 1.0f;  // self-loop
}

__global__ void k_deg_edges(const int* __restrict__ dst, float* __restrict__ deg, int e) {
    int i = blockIdx.x * blockDim.x + threadIdx.x;
    if (i < e) atomicAdd(&deg[dst[i]], 1.0f);
}

__global__ void k_dinv(const float* __restrict__ deg, float* __restrict__ dinv, int n) {
    int i = blockIdx.x * blockDim.x + threadIdx.x;
    if (i < n) {
        float d = deg[i];
        dinv[i] = (d > 0.0f) ? rsqrtf(d) : 0.0f;
    }
}

// ---------------- GEMM1: h1[N,64] = x[N,512] @ W1[512,64]  (bf16 WMMA, f32 acc) ----------------
__global__ void __launch_bounds__(128)
k_gemm1(const float* __restrict__ x, const float* __restrict__ W, float* __restrict__ h, int nrows) {
    // Stage W1 transposed (col-major) as bf16: sW[col][k], 64*512*2B = 64KB LDS.
    __shared__ __bf16 sW[F_HID * F_IN];
    for (int idx = threadIdx.x; idx < F_IN * F_HID; idx += blockDim.x) {
        int k = idx >> 6, c = idx & 63;           // W row-major [k][c]
        sW[c * F_IN + k] = (__bf16)W[idx];
    }
    __syncthreads();

    int wave  = threadIdx.x >> 5;
    int lane  = threadIdx.x & 31;
    int tile  = blockIdx.x * 4 + wave;
    int nTile = (nrows + 15) >> 4;
    if (tile >= nTile) return;

    int row0 = tile << 4;
    int m    = lane & 15;        // row within tile (A), col within n-tile (B/D)
    int hi   = lane >> 4;        // lane-group selector
    int row  = row0 + m; if (row >= nrows) row = nrows - 1;
    const float* arow = x + (size_t)row * F_IN;

    v8f acc[4] = {};             // 4 N-tiles of 16 -> 64 output cols

    #pragma unroll
    for (int ks = 0; ks < F_IN / 32; ++ks) {
        // A fragment: K = [kb, kb+16) contiguous
        int kb = ks * 32 + hi * 16;
        float4 p0 = *(const float4*)(arow + kb);
        float4 p1 = *(const float4*)(arow + kb + 4);
        float4 p2 = *(const float4*)(arow + kb + 8);
        float4 p3 = *(const float4*)(arow + kb + 12);
        v16bf a;
        a[0]=(__bf16)p0.x; a[1]=(__bf16)p0.y; a[2]=(__bf16)p0.z; a[3]=(__bf16)p0.w;
        a[4]=(__bf16)p1.x; a[5]=(__bf16)p1.y; a[6]=(__bf16)p1.z; a[7]=(__bf16)p1.w;
        a[8]=(__bf16)p2.x; a[9]=(__bf16)p2.y; a[10]=(__bf16)p2.z; a[11]=(__bf16)p2.w;
        a[12]=(__bf16)p3.x; a[13]=(__bf16)p3.y; a[14]=(__bf16)p3.z; a[15]=(__bf16)p3.w;

        // B fragments: lane holds col = n*16+m, K = ks*32 + hi*16 + e (contiguous in LDS)
        int kbB = ks * 32 + hi * 16;
        #pragma unroll
        for (int n = 0; n < 4; ++n) {
            v16bf b = *(const v16bf*)(&sW[(size_t)(n * 16 + m) * F_IN + kbB]);
            acc[n] = __builtin_amdgcn_wmma_f32_16x16x32_bf16(
                false, a, false, b, (short)0, acc[n], false, false);
        }
    }

    // D layout: element r -> M = r + hi*8, N = m
    #pragma unroll
    for (int n = 0; n < 4; ++n) {
        int col = n * 16 + m;
        #pragma unroll
        for (int r = 0; r < 8; ++r) {
            int rr = row0 + r + hi * 8;
            if (rr < nrows) h[(size_t)rr * F_HID + col] = acc[n][r];
        }
    }
}

// ---------------- self-loop init: agg1 = h1 * dinv^2 ----------------
__global__ void k_selfloop1(const float* __restrict__ h1, const float* __restrict__ dinv,
                            float* __restrict__ agg1, int total) {
    int t = blockIdx.x * blockDim.x + threadIdx.x;
    if (t < total) {
        int i = t >> 6;
        float dv = dinv[i];
        agg1[t] = h1[t] * dv * dv;
    }
}

// ---------------- edge scatter-add layer 1: one wave32 per edge ----------------
__global__ void k_edge1(const int* __restrict__ src, const int* __restrict__ dst,
                        const float* __restrict__ dinv, const float* __restrict__ h1,
                        float* __restrict__ agg1, int e) {
    int gw   = (blockIdx.x * blockDim.x + threadIdx.x) >> 5;
    int lane = threadIdx.x & 31;
    if (gw >= e) return;
    int s = src[gw], d = dst[gw];
    float w = dinv[s] * dinv[d];
    const float* hs = h1 + (size_t)s * F_HID;
    float* ad = agg1 + (size_t)d * F_HID;
    atomicAdd(&ad[lane],      hs[lane]      * w);
    atomicAdd(&ad[lane + 32], hs[lane + 32] * w);
}

// ---------------- bias + relu -> layer-2 input ----------------
__global__ void k_relu_bias(const float* __restrict__ agg1, const float* __restrict__ b1,
                            float* __restrict__ h2, int total) {
    int t = blockIdx.x * blockDim.x + threadIdx.x;
    if (t < total) {
        float v = agg1[t] + b1[t & 63];
        h2[t] = v > 0.0f ? v : 0.0f;
    }
}

// ---------------- GEMM2: t2[N,10] = h2[N,64] @ W2[64,10]  (N padded to 16) ----------------
__global__ void __launch_bounds__(128)
k_gemm2(const float* __restrict__ h2, const float* __restrict__ W, float* __restrict__ t2, int nrows) {
    __shared__ __bf16 sW[16 * F_HID];      // [col][k], cols 10..15 zero
    for (int idx = threadIdx.x; idx < 16 * F_HID; idx += blockDim.x) {
        int c = idx >> 6, k = idx & 63;
        sW[idx] = (c < F_OUT) ? (__bf16)W[k * F_OUT + c] : (__bf16)0.0f;
    }
    __syncthreads();

    int wave  = threadIdx.x >> 5;
    int lane  = threadIdx.x & 31;
    int tile  = blockIdx.x * 4 + wave;
    int nTile = (nrows + 15) >> 4;
    if (tile >= nTile) return;

    int row0 = tile << 4;
    int m    = lane & 15;
    int hi   = lane >> 4;
    int row  = row0 + m; if (row >= nrows) row = nrows - 1;
    const float* arow = h2 + (size_t)row * F_HID;

    v8f acc = {};
    #pragma unroll
    for (int ks = 0; ks < F_HID / 32; ++ks) {
        int kb = ks * 32 + hi * 16;
        float4 p0 = *(const float4*)(arow + kb);
        float4 p1 = *(const float4*)(arow + kb + 4);
        float4 p2 = *(const float4*)(arow + kb + 8);
        float4 p3 = *(const float4*)(arow + kb + 12);
        v16bf a;
        a[0]=(__bf16)p0.x; a[1]=(__bf16)p0.y; a[2]=(__bf16)p0.z; a[3]=(__bf16)p0.w;
        a[4]=(__bf16)p1.x; a[5]=(__bf16)p1.y; a[6]=(__bf16)p1.z; a[7]=(__bf16)p1.w;
        a[8]=(__bf16)p2.x; a[9]=(__bf16)p2.y; a[10]=(__bf16)p2.z; a[11]=(__bf16)p2.w;
        a[12]=(__bf16)p3.x; a[13]=(__bf16)p3.y; a[14]=(__bf16)p3.z; a[15]=(__bf16)p3.w;

        int kbB = ks * 32 + hi * 16;
        v16bf b = *(const v16bf*)(&sW[(size_t)m * F_HID + kbB]);
        acc = __builtin_amdgcn_wmma_f32_16x16x32_bf16(
            false, a, false, b, (short)0, acc, false, false);
    }

    if (m < F_OUT) {
        #pragma unroll
        for (int r = 0; r < 8; ++r) {
            int rr = row0 + r + hi * 8;
            if (rr < nrows) t2[(size_t)rr * F_OUT + m] = acc[r];
        }
    }
}

// ---------------- self-loop + bias init for output ----------------
__global__ void k_selfloop2(const float* __restrict__ t2, const float* __restrict__ dinv,
                            const float* __restrict__ b2, float* __restrict__ out, int total) {
    int t = blockIdx.x * blockDim.x + threadIdx.x;
    if (t < total) {
        int i = t / F_OUT, c = t - i * F_OUT;
        float dv = dinv[i];
        out[t] = t2[t] * dv * dv + b2[c];
    }
}

// ---------------- edge scatter-add layer 2: 16 lanes per edge (10 active) ----------------
__global__ void k_edge2(const int* __restrict__ src, const int* __restrict__ dst,
                        const float* __restrict__ dinv, const float* __restrict__ t2,
                        float* __restrict__ out, int e) {
    int t = blockIdx.x * blockDim.x + threadIdx.x;
    int ei = t >> 4, c = t & 15;
    if (ei >= e || c >= F_OUT) return;
    int s = src[ei], d = dst[ei];
    float w = dinv[s] * dinv[d];
    atomicAdd(&out[(size_t)d * F_OUT + c], t2[(size_t)s * F_OUT + c] * w);
}

// ---------------- launch ----------------
extern "C" void kernel_launch(void* const* d_in, const int* in_sizes, int n_in,
                              void* d_out, int out_size, void* d_ws, size_t ws_size,
                              hipStream_t stream) {
    const float* x  = (const float*)d_in[0];
    const int*   ei = (const int*)  d_in[1];
    const float* W1 = (const float*)d_in[2];
    const float* b1 = (const float*)d_in[3];
    const float* W2 = (const float*)d_in[4];
    const float* b2 = (const float*)d_in[5];

    const int N = in_sizes[0] / F_IN;
    const int E = in_sizes[1] / 2;
    const int* src = ei;
    const int* dst = ei + E;

    float* ws   = (float*)d_ws;
    float* deg  = ws;                    // N floats
    float* dinv = deg + N;               // N floats
    float* h1   = dinv + N;              // N*64 floats (reused as relu output h2)
    float* agg1 = h1 + (size_t)N * F_HID;// N*64 floats (reused as GEMM2 output t2)
    float* h2   = h1;
    float* t2   = agg1;
    float* out  = (float*)d_out;

    const int TB = 256;
    int nTile  = (N + 15) >> 4;
    int gBlk   = (nTile + 3) >> 2;       // 4 waves (128 thr) per block

    k_init_deg <<<(N + TB - 1) / TB, TB, 0, stream>>>(deg, N);
    k_deg_edges<<<(E + TB - 1) / TB, TB, 0, stream>>>(dst, deg, E);
    k_dinv     <<<(N + TB - 1) / TB, TB, 0, stream>>>(deg, dinv, N);

    k_gemm1    <<<gBlk, 128, 0, stream>>>(x, W1, h1, N);

    int tot1 = N * F_HID;
    k_selfloop1<<<(tot1 + TB - 1) / TB, TB, 0, stream>>>(h1, dinv, agg1, tot1);

    long long e1t = (long long)E * 32;
    k_edge1    <<<(int)((e1t + TB - 1) / TB), TB, 0, stream>>>(src, dst, dinv, h1, agg1, E);

    k_relu_bias<<<(tot1 + TB - 1) / TB, TB, 0, stream>>>(agg1, b1, h2, tot1);

    k_gemm2    <<<gBlk, 128, 0, stream>>>(h2, W2, t2, N);

    int tot2 = N * F_OUT;
    k_selfloop2<<<(tot2 + TB - 1) / TB, TB, 0, stream>>>(t2, dinv, b2, out, tot2);

    long long e2t = (long long)E * 16;
    k_edge2    <<<(int)((e2t + TB - 1) / TB), TB, 0, stream>>>(src, dst, dinv, t2, out, E);
}
